// EdgeUpdate_4028679323808
// MI455X (gfx1250) — matrix operand order repaired and checked by reading, not compiled, not run on hardware
//
#include <hip/hip_runtime.h>

// CDNA5 / gfx1250: wave32, WMMA 16x16x32 f16 -> f32 accumulate.
typedef _Float16 v8h  __attribute__((ext_vector_type(8)));
typedef _Float16 v16h __attribute__((ext_vector_type(16)));
typedef float    v8f  __attribute__((ext_vector_type(8)));

#define HDIM   128
#define INDIM  256
#define TILE_E 32    // two 16-row M-tiles per wave -> 2 independent WMMA chains
#define ROWA   264   // A_lds row stride in halves (256 + 8 pad -> conflict-free b128)
#define ROWH   136   // H_lds row stride in halves (128 + 8 pad)
#define ROWU   132   // U_lds row stride in floats (128 + 4 pad)

// ---------------------------------------------------------------------------
// Pre-pass: convert W1[256,128], W2[128,128] (f32, K-major) into f16,
// transposed to N-major so WMMA B-fragments are two contiguous 16B loads.
// ---------------------------------------------------------------------------
__global__ void convert_weights_kernel(const float* __restrict__ W1,
                                       const float* __restrict__ W2,
                                       _Float16* __restrict__ W1t,
                                       _Float16* __restrict__ W2t) {
  int i = blockIdx.x * blockDim.x + threadIdx.x;
  if (i < HDIM * INDIM) {                       // W1t[n][k] = W1[k][n]
    int n = i >> 8, k = i & 255;
    W1t[i] = (_Float16)W1[k * HDIM + n];
  } else {
    int j = i - HDIM * INDIM;                   // W2t[n][k] = W2[k][n]
    if (j < HDIM * HDIM) {
      int n = j >> 7, k = j & 127;
      W2t[j] = (_Float16)W2[k * HDIM + n];
    }
  }
}

// A-fragment loader: ISA 16-bit A layout -> lane l holds row M=l&15,
// halves {k..k+7, k+16..k+23} with k = 32*kb + 8*(l>>4).
__device__ __forceinline__ v16h load_a_frag(const _Float16* p) {
  v8h lo = *(const v8h*)p;
  v8h hh = *(const v8h*)(p + 16);
  return __builtin_shufflevector(lo, hh, 0,1,2,3,4,5,6,7,8,9,10,11,12,13,14,15);
}

// ---------------------------------------------------------------------------
// Main fused kernel: gather -> GEMM1(+bias,ReLU) -> GEMM2(+bias) -> LayerNorm
// -> residual add.  256 threads = 8 waves; wave w owns output cols [16w,16w+16)
// for BOTH 16-row M-tiles of the 32-edge tile.  B fragments for both GEMMs
// live in VGPRs across the whole grid-stride tile loop.  A-fragment loads are
// software-pipelined one k-block ahead of the WMMAs that consume them.
// ---------------------------------------------------------------------------
__launch_bounds__(256)
__global__ void edge_update_kernel(const float* __restrict__ node_inv,
                                   const float* __restrict__ edge_features,
                                   const int*   __restrict__ edge_src,
                                   const int*   __restrict__ edge_dst,
                                   const float* __restrict__ b1,
                                   const float* __restrict__ b2,
                                   const float* __restrict__ gamma_p,
                                   const float* __restrict__ beta_p,
                                   const _Float16* __restrict__ W1t,
                                   const _Float16* __restrict__ W2t,
                                   float* __restrict__ out,
                                   int n_tiles) {
  __shared__ __align__(16) _Float16 A_lds[TILE_E * ROWA];   // 16.5 KB
  __shared__ __align__(16) _Float16 H_lds[TILE_E * ROWH];   //  8.5 KB
  __shared__ __align__(16) float    U_lds[TILE_E * ROWU];   // 16.5 KB

  const int tid  = threadIdx.x;
  const int lane = tid & 31;
  const int w    = tid >> 5;      // wave id 0..7
  const int l16  = lane & 15;
  const int hi   = lane >> 4;

  // ---- persistent WMMA B fragments (lane = N col, halves = {k..k+7, k+16..k+23}, k = 8*hi) ----
  v16h B1[8];                     // GEMM1: K = 256 -> 8 k-blocks of 32
  v16h B2[4];                     // GEMM2: K = 128 -> 4 k-blocks of 32
  {
    const int rb = w * 16 + l16;  // output column owned by this lane
    #pragma unroll
    for (int kb = 0; kb < 8; ++kb)
      B1[kb] = load_a_frag(W1t + rb * INDIM + kb * 32 + hi * 8);
    #pragma unroll
    for (int kb = 0; kb < 4; ++kb)
      B2[kb] = load_a_frag(W2t + rb * HDIM + kb * 32 + hi * 8);
  }
  const float bias1 = b1[w * 16 + l16];
  const float bias2 = b2[w * 16 + l16];
  float gma[8], bta[8];
  {
    const int c = tid & 15;
    #pragma unroll
    for (int j = 0; j < 8; ++j) { gma[j] = gamma_p[c * 8 + j]; bta[j] = beta_p[c * 8 + j]; }
  }

  const _Float16* arow0 = &A_lds[l16 * ROWA + hi * 8];
  const _Float16* arow1 = &A_lds[(l16 + 16) * ROWA + hi * 8];
  const _Float16* hrow0 = &H_lds[l16 * ROWH + hi * 8];
  const _Float16* hrow1 = &H_lds[(l16 + 16) * ROWH + hi * 8];

  for (int t = blockIdx.x; t < n_tiles; t += gridDim.x) {
    const long e0 = (long)t * TILE_E;

    // ---- stage A tile [32 x 256] f16 into LDS: gather + concat + convert ----
    #pragma unroll 4
    for (int j = 0; j < TILE_E; ++j) {
      const long e = e0 + j;
      float v;
      if (tid < 64)       v = node_inv[(long)edge_src[e] * 64 + tid];
      else if (tid < 128) v = node_inv[(long)edge_dst[e] * 64 + (tid - 64)];
      else                v = edge_features[e * HDIM + (tid - 128)];
      A_lds[j * ROWA + tid] = (_Float16)v;
    }
    __syncthreads();

    // ---- GEMM1: two independent accumulators, loads pipelined 1 kb ahead ----
    v8f acc0, acc1;
    #pragma unroll
    for (int i = 0; i < 8; ++i) { acc0[i] = bias1; acc1[i] = bias1; }
    {
      v16h a0 = load_a_frag(arow0);
      v16h a1 = load_a_frag(arow1);
      #pragma unroll
      for (int kb = 0; kb < 8; ++kb) {
        v16h n0 = a0, n1 = a1;
        if (kb < 7) {                         // constant-folded after unroll
          n0 = load_a_frag(arow0 + (kb + 1) * 32);
          n1 = load_a_frag(arow1 + (kb + 1) * 32);
        }
        acc0 = __builtin_amdgcn_wmma_f32_16x16x32_f16(false, a0, false, B1[kb],
                                                      (short)0, acc0, false, false);
        acc1 = __builtin_amdgcn_wmma_f32_16x16x32_f16(false, a1, false, B1[kb],
                                                      (short)0, acc1, false, false);
        a0 = n0; a1 = n1;
      }
    }
    // bias already folded; fused ReLU, convert to f16, store H tile
    #pragma unroll
    for (int v = 0; v < 8; ++v) {   // C layout: vgpr v -> M = v + 8*hi, lane -> N
      const int m = v + hi * 8;
      H_lds[m * ROWH + w * 16 + l16]        = (_Float16)__builtin_fmaxf(acc0[v], 0.f);
      H_lds[(m + 16) * ROWH + w * 16 + l16] = (_Float16)__builtin_fmaxf(acc1[v], 0.f);
    }
    __syncthreads();

    // ---- GEMM2: [32x128] x [128x16] per wave, bias fused, pipelined ----
    v8f d0, d1;
    #pragma unroll
    for (int i = 0; i < 8; ++i) { d0[i] = bias2; d1[i] = bias2; }
    {
      v16h a0 = load_a_frag(hrow0);
      v16h a1 = load_a_frag(hrow1);
      #pragma unroll
      for (int kb = 0; kb < 4; ++kb) {
        v16h n0 = a0, n1 = a1;
        if (kb < 3) {
          n0 = load_a_frag(hrow0 + (kb + 1) * 32);
          n1 = load_a_frag(hrow1 + (kb + 1) * 32);
        }
        d0 = __builtin_amdgcn_wmma_f32_16x16x32_f16(false, a0, false, B2[kb],
                                                    (short)0, d0, false, false);
        d1 = __builtin_amdgcn_wmma_f32_16x16x32_f16(false, a1, false, B2[kb],
                                                    (short)0, d1, false, false);
        a0 = n0; a1 = n1;
      }
    }
    #pragma unroll
    for (int v = 0; v < 8; ++v) {
      const int m = v + hi * 8;
      U_lds[m * ROWU + w * 16 + l16]        = d0[v];
      U_lds[(m + 16) * ROWU + w * 16 + l16] = d1[v];
    }
    __syncthreads();

    // ---- LayerNorm over 128 ch per edge + residual; 16 threads per row,
    //      two row-halves per thread (32 rows total) ----
    {
      const int r0 = tid >> 4, c = tid & 15;
      #pragma unroll
      for (int half = 0; half < 2; ++half) {
        const int r = r0 + half * 16;
        const float* up = &U_lds[r * ROWU + c * 8];
        float u[8], s1 = 0.f, s2 = 0.f;
        #pragma unroll
        for (int j = 0; j < 8; ++j) { u[j] = up[j]; s1 += u[j]; s2 += u[j] * u[j]; }
        #pragma unroll
        for (int m = 8; m >= 1; m >>= 1) {      // reduce within 16-lane group
          s1 += __shfl_xor(s1, m, 32);
          s2 += __shfl_xor(s2, m, 32);
        }
        const float mu  = s1 * (1.f / 128.f);
        const float var = s2 * (1.f / 128.f) - mu * mu;
        const float rs  = rsqrtf(var + 1e-5f);
        const long  e   = e0 + r;
        #pragma unroll
        for (int j = 0; j < 8; ++j) {
          const int k = c * 8 + j;
          out[e * HDIM + k] = edge_features[e * HDIM + k] + (u[j] - mu) * rs * gma[j] + bta[j];
        }
      }
    }
    __syncthreads();   // protect LDS before next tile's staging
  }
}

// ---------------------------------------------------------------------------
extern "C" void kernel_launch(void* const* d_in, const int* in_sizes, int n_in,
                              void* d_out, int out_size, void* d_ws, size_t ws_size,
                              hipStream_t stream) {
  const float* node_inv      = (const float*)d_in[0];
  const float* edge_features = (const float*)d_in[1];
  const int*   edge_src      = (const int*)d_in[2];
  const int*   edge_dst      = (const int*)d_in[3];
  const float* W1            = (const float*)d_in[4];
  const float* b1            = (const float*)d_in[5];
  const float* W2            = (const float*)d_in[6];
  const float* b2            = (const float*)d_in[7];
  const float* gamma_p       = (const float*)d_in[8];
  const float* beta_p        = (const float*)d_in[9];
  float* out = (float*)d_out;

  _Float16* W1t = (_Float16*)d_ws;          // 128*256 f16 = 64 KB
  _Float16* W2t = W1t + HDIM * INDIM;       // 128*128 f16 = 32 KB

  const int n_edges = in_sizes[2];
  const int n_tiles = n_edges / TILE_E;     // 800000/32 = 25000, exact

  const int conv_elems = HDIM * INDIM + HDIM * HDIM;
  convert_weights_kernel<<<(conv_elems + 255) / 256, 256, 0, stream>>>(W1, W2, W1t, W2t);

  int grid = n_tiles < 2048 ? n_tiles : 2048;  // grid-stride: amortize B-frag loads
  edge_update_kernel<<<grid, 256, 0, stream>>>(node_inv, edge_features, edge_src, edge_dst,
                                               b1, b2, gamma_p, beta_p, W1t, W2t, out, n_tiles);
}
